// nha_layer_36189394436457
// MI455X (gfx1250) — matrix operand order repaired and checked by reading, not compiled, not run on hardware
//
#include <hip/hip_runtime.h>

// ---------------------------------------------------------------- types
typedef __bf16 bf16_t;
typedef __attribute__((ext_vector_type(8)))  __bf16 v8bf;
typedef __attribute__((ext_vector_type(16))) __bf16 v16bf;
typedef __attribute__((ext_vector_type(8)))  float  v8f;

typedef unsigned short u16;
typedef unsigned int   u32;

// CDNA5 async global->LDS path (guarded: falls back to reg-staged copies)
#if __has_builtin(__builtin_amdgcn_global_load_async_to_lds_b128) && \
    __has_builtin(__builtin_amdgcn_s_wait_asynccnt)
#define USE_ASYNC_LDS 1
#else
#define USE_ASYNC_LDS 0
#endif

// pointer types matching the builtin signature: v4i in AS1 (global) / AS3 (LDS)
typedef int v4i_vs __attribute__((vector_size(16)));
typedef __attribute__((address_space(1))) v4i_vs* as1_v4i_p;
typedef __attribute__((address_space(3))) v4i_vs* as3_v4i_p;

static __device__ __forceinline__ void async_copy_b128(const void* gp, void* lp) {
#if USE_ASYNC_LDS
    __builtin_amdgcn_global_load_async_to_lds_b128((as1_v4i_p)gp, (as3_v4i_p)lp, 0, 0);
#else
    (void)gp; (void)lp;
#endif
}

// problem dims
#define B_   2
#define N_   2048
#define NH_  32
#define EIN_ 128
#define MD_  512
#define H_   8
#define HD_  64
#define FF_  2048
#define OUT_ 128
#define PED_ 32
#define TAB_ 512
#define ROWS_ 131072L   // B*N*NH
#define M_    4096L     // B*N

static __device__ __forceinline__ u16 f2bf(float f) {
    u32 u = __builtin_bit_cast(u32, f);
    u += 0x7FFFu + ((u >> 16) & 1u);            // round-to-nearest-even
    return (u16)(u >> 16);
}
static __device__ __forceinline__ bf16_t bfc(u16 r) {
    return __builtin_bit_cast(bf16_t, r);
}

// ---------------------------------------------------------------- f32 -> bf16 convert (x4)
__global__ __launch_bounds__(256) void cvt_kernel(const float* __restrict__ src,
                                                  u16* __restrict__ dst, long n) {
    long i = ((long)blockIdx.x * 256 + threadIdx.x) * 4;
    if (i + 3 < n) {
        float4 f = *(const float4*)(src + i);
        u16 r0 = f2bf(f.x), r1 = f2bf(f.y), r2 = f2bf(f.z), r3 = f2bf(f.w);
        u32 lo = (u32)r0 | ((u32)r1 << 16);
        u32 hi = (u32)r2 | ((u32)r3 << 16);
        *(uint2*)(dst + i) = make_uint2(lo, hi);
    } else {
        for (; i < n; ++i) dst[i] = f2bf(src[i]);
    }
}

// ---------------------------------------------------------------- tiled bf16 WMMA GEMM
// C[M,N] (f32 accum) = A[M,K]_bf16 @ B[K,N]_bf16 ; BM=BN=128, BK=32
// 8 waves: wave w -> wm = w&3 (32-row strip), wn = w>>2 (64-col strip)
// Double-buffered LDS; A tile staged with GLOBAL_LOAD_ASYNC_TO_LDS_B128 when available.
// ACT: 0 none, 1 SiLU on f32 accumulator before store.
#define BM 128
#define BN 128
#define BK 32
template<int ACT, bool WF32, bool WBF16>
__global__ __launch_bounds__(256)
void gemm_bf16(const u16* __restrict__ A, const u16* __restrict__ B,
               float* __restrict__ Cf, u16* __restrict__ Cb,
               int M, int N, int K)
{
    __shared__ u16 As[2][BM * BK];    // row-major [128][32]
    __shared__ u16 Bs[2][BN * BK];    // TRANSPOSED: [n][k] = [128][32]

    const int tid  = threadIdx.x;
    const int lane = tid & 31;
    const int w    = tid >> 5;
    const int wm   = w & 3;
    const int wn   = w >> 2;
    const int hi   = lane >> 4;       // lane half (wave32)
    const int ln   = lane & 15;
    const long bm  = (long)blockIdx.y * BM;
    const long bn  = (long)blockIdx.x * BN;

    // stage A tile (direct copy, 512 x 16B chunks, 2 per thread)
    auto stageA = [&](int k0, int buf) {
#pragma unroll
        for (int c = 0; c < 2; ++c) {
            int cid = tid + c * 256;
            int row = cid >> 2;
            int ko  = (cid & 3) << 3;
            const u16* gp = A + (bm + row) * (long)K + k0 + ko;
            u16*       lp = &As[buf][row * BK + ko];
#if USE_ASYNC_LDS
            async_copy_b128(gp, lp);
#else
            *(v8bf*)lp = *(const v8bf*)gp;
#endif
        }
    };
    // stage B tile transposed: read 8 contiguous along N, scatter along K
    auto stageB = [&](int k0, int buf) {
#pragma unroll
        for (int c = 0; c < 2; ++c) {
            int cid = tid + c * 256;
            int kk  = cid >> 4;               // 0..31
            int no  = (cid & 15) << 3;        // 0..120
            v8bf t = *(const v8bf*)(B + (long)(k0 + kk) * N + bn + no);
#pragma unroll
            for (int i = 0; i < 8; ++i)
                Bs[buf][(no + i) * BK + kk] = ((const u16*)&t)[i];
        }
    };

    v8f acc[2][4];
#pragma unroll
    for (int mt = 0; mt < 2; ++mt)
#pragma unroll
        for (int nt = 0; nt < 4; ++nt) acc[mt][nt] = (v8f)0.0f;

    stageA(0, 0);
    stageB(0, 0);
    int cur = 0;

    for (int k0 = 0; k0 < K; k0 += BK) {
#if USE_ASYNC_LDS
        __builtin_amdgcn_s_wait_asynccnt(0);
#endif
        __syncthreads();                       // tile `cur` resident; prev reads done
        if (k0 + BK < K) {                     // overlap: stage next tile
            stageA(k0 + BK, cur ^ 1);
            stageB(k0 + BK, cur ^ 1);
        }

        // ---- A fragments (ISA 16-bit A layout: halves {b..b+7, b+16..b+23}, b=8*hi)
        v16bf af[2];
        const int ab = hi * 8;
#pragma unroll
        for (int mt = 0; mt < 2; ++mt) {
            int r = wm * 32 + mt * 16 + ln;
            v8bf lo = *(const v8bf*)(&As[cur][r * BK + ab]);
            v8bf hb = *(const v8bf*)(&As[cur][r * BK + ab + 16]);
            af[mt] = __builtin_shufflevector(lo, hb, 0,1,2,3,4,5,6,7,8,9,10,11,12,13,14,15);
        }
        // ---- B fragments (lane holds col=ln, K halves {b..b+7, b+16..b+23}, b=8*hi)
        v16bf bf[4];
        const int kb = hi * 8;
#pragma unroll
        for (int nt = 0; nt < 4; ++nt) {
            int cc = wn * 64 + nt * 16 + ln;
            v8bf lo = *(const v8bf*)(&Bs[cur][cc * BK + kb]);
            v8bf hb = *(const v8bf*)(&Bs[cur][cc * BK + kb + 16]);
            bf[nt] = __builtin_shufflevector(lo, hb, 0,1,2,3,4,5,6,7,8,9,10,11,12,13,14,15);
        }
#pragma unroll
        for (int mt = 0; mt < 2; ++mt)
#pragma unroll
            for (int nt = 0; nt < 4; ++nt)
                acc[mt][nt] = __builtin_amdgcn_wmma_f32_16x16x32_bf16(
                    false, af[mt], false, bf[nt], (short)0, acc[mt][nt], false, false);
        cur ^= 1;
    }

    // ---- epilogue: C layout M = g + 8*hi, N = ln
#pragma unroll
    for (int mt = 0; mt < 2; ++mt)
#pragma unroll
        for (int nt = 0; nt < 4; ++nt)
#pragma unroll
            for (int g = 0; g < 8; ++g) {
                long row = bm + wm * 32 + mt * 16 + g + hi * 8;
                long col = bn + wn * 64 + nt * 16 + ln;
                float v = acc[mt][nt][g];
                if (ACT == 1) v = v / (1.0f + __expf(-v));   // SiLU
                if (WF32)  Cf[row * (long)N + col] = v;
                if (WBF16) Cb[row * (long)N + col] = f2bf(v);
            }
}

// ---------------------------------------------------------------- relative position bias
// bias[m][h][i][j] = sum_p dtab[d[m,i,j]][p]*ptab[phi[m,i,j]][p] * Wb[p][h]
__global__ __launch_bounds__(256)
void bias_kernel(const int* __restrict__ dist, const int* __restrict__ phi,
                 const float* __restrict__ dtab, const float* __restrict__ ptab,
                 const float* __restrict__ wb, float* __restrict__ bias)
{
    __shared__ float swb[PED_ * H_];          // 32*8
    int tid = threadIdx.x;
    long m  = blockIdx.x;
    if (tid < PED_ * H_) swb[tid] = wb[tid];
    __syncthreads();
#pragma unroll
    for (int it = 0; it < 4; ++it) {
        int pidx = tid + it * 256;            // 0..1023 -> (i,j)
        int i = pidx >> 5, j = pidx & 31;
        long idx = m * 1024 + pidx;
        int d = dist[idx], ph = phi[idx];
        const float* dr = dtab + (long)d * PED_;
        const float* pr = ptab + (long)ph * PED_;
        float acc[H_];
#pragma unroll
        for (int hh = 0; hh < H_; ++hh) acc[hh] = 0.f;
        for (int p = 0; p < PED_; ++p) {
            float pe = dr[p] * pr[p];
#pragma unroll
            for (int hh = 0; hh < H_; ++hh) acc[hh] += pe * swb[p * H_ + hh];
        }
#pragma unroll
        for (int hh = 0; hh < H_; ++hh)
            bias[((m * H_ + hh) * NH_ + i) * NH_ + j] = acc[hh];
    }
}

// ---------------------------------------------------------------- attention: 1 wave = 1 (m,h)
// S = (Q Kt)/8 + bias ; P = softmax(S) ; O = P V    (nh=32, hd=64)
__global__ __launch_bounds__(128)
void attn_kernel(const u16* __restrict__ Q, const u16* __restrict__ Kb,
                 const u16* __restrict__ V, const float* __restrict__ bias,
                 u16* __restrict__ O)
{
    __shared__ u16 P[4][32][40];              // per-wave P tile, padded to 40 halves
    const int lane = threadIdx.x & 31;
    const int wv   = threadIdx.x >> 5;
    const long gw  = (long)blockIdx.x * 4 + wv;
    const long m   = gw >> 3;
    const int  h   = (int)(gw & 7);
    const int  hi  = lane >> 4, ln = lane & 15;
    const long base = (m * NH_) * (long)MD_ + h * HD_;   // element offset of row0

    // Q A-fragments [mt][kt]
    v16bf qa[2][2];
#pragma unroll
    for (int mt = 0; mt < 2; ++mt)
#pragma unroll
        for (int kt = 0; kt < 2; ++kt) {
            long off = base + (long)(mt * 16 + ln) * MD_ + kt * 32 + hi * 8;
            v8bf lo = *(const v8bf*)(Q + off);
            v8bf hb = *(const v8bf*)(Q + off + 16);
            qa[mt][kt] = __builtin_shufflevector(lo, hb, 0,1,2,3,4,5,6,7,8,9,10,11,12,13,14,15);
        }
    // K^T B-fragments [nt][kt]: lane -> column j = nt*16+ln, d halves {8*hi..+7, 8*hi+16..+23}
    v16bf kf[2][2];
#pragma unroll
    for (int nt = 0; nt < 2; ++nt)
#pragma unroll
        for (int kt = 0; kt < 2; ++kt) {
            long off = base + (long)(nt * 16 + ln) * MD_ + kt * 32 + hi * 8;
            v8bf lo = *(const v8bf*)(Kb + off);
            v8bf hb = *(const v8bf*)(Kb + off + 16);
            kf[nt][kt] = __builtin_shufflevector(lo, hb, 0,1,2,3,4,5,6,7,8,9,10,11,12,13,14,15);
        }
    // scores
    v8f s[2][2];
#pragma unroll
    for (int mt = 0; mt < 2; ++mt)
#pragma unroll
        for (int nt = 0; nt < 2; ++nt) {
            v8f z = (v8f)0.0f;
            z = __builtin_amdgcn_wmma_f32_16x16x32_bf16(false, qa[mt][0], false, kf[nt][0], (short)0, z, false, false);
            z = __builtin_amdgcn_wmma_f32_16x16x32_bf16(false, qa[mt][1], false, kf[nt][1], (short)0, z, false, false);
#pragma unroll
            for (int g = 0; g < 8; ++g) {
                int i = mt * 16 + g + hi * 8;
                int j = nt * 16 + ln;
                z[g] = z[g] * 0.125f + bias[((m * H_ + h) * NH_ + i) * NH_ + j];
            }
            s[mt][nt] = z;
        }
    // row softmax (row lives across 16 lanes of one half-wave + 2 N-tiles)
#pragma unroll
    for (int mt = 0; mt < 2; ++mt)
#pragma unroll
        for (int g = 0; g < 8; ++g) {
            float a0 = s[mt][0][g], a1 = s[mt][1][g];
            float mx = fmaxf(a0, a1);
            for (int o = 8; o >= 1; o >>= 1) mx = fmaxf(mx, __shfl_xor(mx, o, 16));
            float e0 = __expf(a0 - mx), e1 = __expf(a1 - mx);
            float sm = e0 + e1;
            for (int o = 8; o >= 1; o >>= 1) sm += __shfl_xor(sm, o, 16);
            float inv = 1.0f / sm;
            s[mt][0][g] = e0 * inv;
            s[mt][1][g] = e1 * inv;
        }
    // P -> LDS (C layout -> row-major), then reload as A-fragments
#pragma unroll
    for (int mt = 0; mt < 2; ++mt)
#pragma unroll
        for (int nt = 0; nt < 2; ++nt)
#pragma unroll
            for (int g = 0; g < 8; ++g)
                P[wv][mt * 16 + g + hi * 8][nt * 16 + ln] = f2bf(s[mt][nt][g]);
    __syncthreads();

    v16bf pa[2];
#pragma unroll
    for (int mt = 0; mt < 2; ++mt) {
        int r = mt * 16 + ln;
        v8bf lo = *(const v8bf*)(&P[wv][r][hi * 8]);
        v8bf hb = *(const v8bf*)(&P[wv][r][hi * 8 + 16]);
        pa[mt] = __builtin_shufflevector(lo, hb, 0,1,2,3,4,5,6,7,8,9,10,11,12,13,14,15);
    }
    // V B-fragments [nt]: lane col n = nt*16+ln, K rows {8*hi..+7, 8*hi+16..+23} (stride MD_)
    v16bf vf[4];
#pragma unroll
    for (int nt = 0; nt < 4; ++nt) {
        int cc = nt * 16 + ln;
        int kk0 = hi * 8;
        v16bf t;
#pragma unroll
        for (int t2 = 0; t2 < 16; ++t2)
            t[t2] = bfc(V[base + (long)(kk0 + t2 + (t2 >= 8 ? 8 : 0)) * MD_ + cc]);
        vf[nt] = t;
    }
    // O = P V, write bf16
#pragma unroll
    for (int mt = 0; mt < 2; ++mt)
#pragma unroll
        for (int nt = 0; nt < 4; ++nt) {
            v8f o = (v8f)0.0f;
            o = __builtin_amdgcn_wmma_f32_16x16x32_bf16(false, pa[mt], false, vf[nt], (short)0, o, false, false);
#pragma unroll
            for (int g = 0; g < 8; ++g) {
                int i = mt * 16 + g + hi * 8;
                int n = nt * 16 + ln;
                O[base + (long)i * MD_ + n] = f2bf(o[g]);
            }
        }
}

// ---------------------------------------------------------------- fused add + LayerNorm
// Y = LN(X + R) * g + b ; 1 wave per 512-wide row, 16 elems/lane
template<bool WF32, bool WBF16>
__global__ __launch_bounds__(256)
void add_ln_kernel(const float* __restrict__ X, const float* __restrict__ R,
                   const float* __restrict__ g, const float* __restrict__ b,
                   float* __restrict__ Yf, u16* __restrict__ Yb)
{
    const int lane = threadIdx.x & 31;
    const int wv   = threadIdx.x >> 5;
    const long row = (long)blockIdx.x * 8 + wv;
    const float* xr = X + row * MD_;
    const float* rr = R + row * MD_;
    float v[16];
    float s = 0.f;
#pragma unroll
    for (int t = 0; t < 4; ++t) {
        int c = lane * 16 + t * 4;
        float4 a = *(const float4*)(xr + c);
        float4 d = *(const float4*)(rr + c);
        v[t*4+0] = a.x + d.x; v[t*4+1] = a.y + d.y;
        v[t*4+2] = a.z + d.z; v[t*4+3] = a.w + d.w;
        s += v[t*4+0] + v[t*4+1] + v[t*4+2] + v[t*4+3];
    }
    for (int o = 16; o >= 1; o >>= 1) s += __shfl_xor(s, o, 32);
    float mean = s * (1.0f / MD_);
    float vs = 0.f;
#pragma unroll
    for (int t = 0; t < 16; ++t) { float d = v[t] - mean; vs += d * d; }
    for (int o = 16; o >= 1; o >>= 1) vs += __shfl_xor(vs, o, 32);
    float inv = rsqrtf(vs * (1.0f / MD_) + 1e-5f);
#pragma unroll
    for (int t = 0; t < 16; ++t) {
        int c = lane * 16 + t;
        float y = (v[t] - mean) * inv * g[c] + b[c];
        if (WF32)  Yf[row * MD_ + c] = y;
        if (WBF16) Yb[row * MD_ + c] = f2bf(y);
    }
}

// ---------------------------------------------------------------- launcher
extern "C" void kernel_launch(void* const* d_in, const int* in_sizes, int n_in,
                              void* d_out, int out_size, void* d_ws, size_t ws_size,
                              hipStream_t stream)
{
    (void)in_sizes; (void)n_in; (void)out_size; (void)ws_size;
    const float* x      = (const float*)d_in[0];
    const int*   dists  = (const int*)d_in[1];
    const int*   phis   = (const int*)d_in[2];
    const float* W_in   = (const float*)d_in[3];
    const float* dtab   = (const float*)d_in[4];
    const float* ptab   = (const float*)d_in[5];
    const float* W_bias = (const float*)d_in[6];
    const float* Wq     = (const float*)d_in[7];
    const float* Wk     = (const float*)d_in[8];
    const float* Wv     = (const float*)d_in[9];
    const float* Wo     = (const float*)d_in[10];
    const float* Wff1   = (const float*)d_in[11];
    const float* Wff2   = (const float*)d_in[12];
    const float* g1     = (const float*)d_in[13];
    const float* b1     = (const float*)d_in[14];
    const float* g2     = (const float*)d_in[15];
    const float* b2     = (const float*)d_in[16];
    const float* W_out  = (const float*)d_in[17];
    float* out = (float*)d_out;

    char* p = (char*)d_ws;
    auto carve = [&](size_t bytes) -> char* {
        char* r = p; p += (bytes + 255) & ~(size_t)255; return r;
    };
    u16*   xb    = (u16*)  carve(ROWS_ * EIN_ * 2);
    u16*   Winb  = (u16*)  carve(EIN_ * MD_ * 2);
    u16*   Wqb   = (u16*)  carve(MD_ * MD_ * 2);
    u16*   Wkb   = (u16*)  carve(MD_ * MD_ * 2);
    u16*   Wvb   = (u16*)  carve(MD_ * MD_ * 2);
    u16*   Wob   = (u16*)  carve(MD_ * MD_ * 2);
    u16*   Wf1b  = (u16*)  carve(MD_ * FF_ * 2);
    u16*   Wf2b  = (u16*)  carve(FF_ * MD_ * 2);
    u16*   Woutb = (u16*)  carve(MD_ * OUT_ * 2);
    float* hf    = (float*)carve(ROWS_ * MD_ * 4);
    u16*   hb    = (u16*)  carve(ROWS_ * MD_ * 2);
    u16*   qb    = (u16*)  carve(ROWS_ * MD_ * 2);
    u16*   kb    = (u16*)  carve(ROWS_ * MD_ * 2);
    u16*   vb    = (u16*)  carve(ROWS_ * MD_ * 2);
    float* biasb = (float*)carve(M_ * H_ * NH_ * NH_ * 4);
    u16*   ob    = (u16*)  carve(ROWS_ * MD_ * 2);
    float* attnf = (float*)carve(ROWS_ * MD_ * 4);
    float* h1f   = (float*)carve(ROWS_ * MD_ * 4);
    u16*   h1b   = (u16*)  carve(ROWS_ * MD_ * 2);
    u16*   ffsb  = (u16*)  carve(ROWS_ * FF_ * 2);
    float* ff2f  = (float*)carve(ROWS_ * MD_ * 4);
    u16*   h2b   = (u16*)  carve(ROWS_ * MD_ * 2);

    auto cvt = [&](const float* s, u16* d, long n) {
        long thr = (n + 3) / 4;
        cvt_kernel<<<dim3((unsigned)((thr + 255) / 256)), dim3(256), 0, stream>>>(s, d, n);
    };
    // 1) bf16 conversions
    cvt(x,     xb,    ROWS_ * EIN_);
    cvt(W_in,  Winb,  EIN_ * MD_);
    cvt(Wq,    Wqb,   MD_ * MD_);
    cvt(Wk,    Wkb,   MD_ * MD_);
    cvt(Wv,    Wvb,   MD_ * MD_);
    cvt(Wo,    Wob,   MD_ * MD_);
    cvt(Wff1,  Wf1b,  MD_ * FF_);
    cvt(Wff2,  Wf2b,  FF_ * MD_);
    cvt(W_out, Woutb, MD_ * OUT_);

    const dim3 blk(256);
    // 2) h = x @ W_in   (f32 + bf16)
    gemm_bf16<0, true, true><<<dim3(MD_ / BN, ROWS_ / BM), blk, 0, stream>>>(
        xb, Winb, hf, hb, (int)ROWS_, MD_, EIN_);
    // 3) q,k,v projections (bf16)
    gemm_bf16<0, false, true><<<dim3(MD_ / BN, ROWS_ / BM), blk, 0, stream>>>(
        hb, Wqb, nullptr, qb, (int)ROWS_, MD_, MD_);
    gemm_bf16<0, false, true><<<dim3(MD_ / BN, ROWS_ / BM), blk, 0, stream>>>(
        hb, Wkb, nullptr, kb, (int)ROWS_, MD_, MD_);
    gemm_bf16<0, false, true><<<dim3(MD_ / BN, ROWS_ / BM), blk, 0, stream>>>(
        hb, Wvb, nullptr, vb, (int)ROWS_, MD_, MD_);
    // 4) relative position bias
    bias_kernel<<<dim3((unsigned)M_), blk, 0, stream>>>(dists, phis, dtab, ptab, W_bias, biasb);
    // 5) attention (WMMA per head)
    attn_kernel<<<dim3((unsigned)(M_ * H_ / 4)), dim3(128), 0, stream>>>(qb, kb, vb, biasb, ob);
    // 6) att_out = o @ Wo (f32)
    gemm_bf16<0, true, false><<<dim3(MD_ / BN, ROWS_ / BM), blk, 0, stream>>>(
        ob, Wob, attnf, nullptr, (int)ROWS_, MD_, MD_);
    // 7) h1 = LN(h + att_out)
    add_ln_kernel<true, true><<<dim3((unsigned)(ROWS_ / 8)), blk, 0, stream>>>(
        hf, attnf, g1, b1, h1f, h1b);
    // 8) ff = silu(h1 @ Wff1) (bf16, fused SiLU)
    gemm_bf16<1, false, true><<<dim3(FF_ / BN, ROWS_ / BM), blk, 0, stream>>>(
        h1b, Wf1b, nullptr, ffsb, (int)ROWS_, FF_, MD_);
    // 9) ff2 = ff @ Wff2 (f32)
    gemm_bf16<0, true, false><<<dim3(MD_ / BN, ROWS_ / BM), blk, 0, stream>>>(
        ffsb, Wf2b, ff2f, nullptr, (int)ROWS_, MD_, FF_);
    // 10) h2 = LN(h1 + ff2) (bf16)
    add_ln_kernel<false, true><<<dim3((unsigned)(ROWS_ / 8)), blk, 0, stream>>>(
        h1f, ff2f, g2, b2, nullptr, h2b);
    // 11) out = h2 @ W_out (f32 straight to d_out)
    gemm_bf16<0, true, false><<<dim3(OUT_ / BN, ROWS_ / BM), blk, 0, stream>>>(
        h2b, Woutb, out, nullptr, (int)ROWS_, OUT_, MD_);
}
